// ExactLTCLayer_38594576122137
// MI455X (gfx1250) — matrix operand order repaired and checked by reading, not compiled, not run on hardware
//
#include <hip/hip_runtime.h>

typedef _Float16 v16h __attribute__((ext_vector_type(16)));
typedef _Float16 v2h  __attribute__((ext_vector_type(2)));
typedef float    v8f  __attribute__((ext_vector_type(8)));

#define UNITS  256
#define IDIM   64
#define BATCH  128
#define TSTEPS 1024
#define LOG2E  1.4426950408889634f

// One wave handles: 1 unit u, 8 batches x 2 timesteps in the 16 WMMA columns,
// looping over t (the only sequential dimension). The D=64 reductions
// (sum f, sum A*f) run on the matrix pipe:
//   A-matrix (16x32 f16): row0 = ones, row1 = A[u,:], rows 2..15 = 0
//   B-matrix (32x16 f16): f = sigmoid(sigma*(inp-mu)), k = d, n = (b,ts)
//   D rows 0/1 => fs and sum(A*f) in c[0], c[1] of every column lane.
__global__ __launch_bounds__(256) void ltc_scan_wmma(
    const float* __restrict__ inp,   // [B, T, D]
    const float* __restrict__ Aw,    // [U, D]
    const float* __restrict__ Sg,    // [U, D]
    const float* __restrict__ Mu,    // [U, D]
    const float* __restrict__ X0,    // [U]
    float* __restrict__ out)         // [B, T, U]
{
    const int lane  = threadIdx.x & 31;
    const int wid   = (blockIdx.x * blockDim.x + threadIdx.x) >> 5; // 0..4095
    const int u     = wid & (UNITS - 1);
    const int btile = wid >> 8;               // 0..15 (8 batches each)
    const int n     = lane & 15;              // WMMA column / row index
    const int hi    = lane >> 4;              // lane half (k-split)
    const int bb    = btile * 8 + (n & 7);    // this column's batch
    const int tsel  = n >> 3;                 // 0/1: timestep within pair

    // ---- per-wave setup, hoisted out of the whole T loop ----------------
    // sigmoid(sigma*(x-mu)) = rcp(1 + exp2(c0 - sigL*x)),
    //   sigL = sigma*log2(e), c0 = sigL*mu
    float sgr[32], c0r[32];   // index [frag*16 + j], d = frag*32 + hi*16 + j
#pragma unroll
    for (int frag = 0; frag < 2; ++frag) {
        const float* sp = Sg + u * IDIM + frag * 32 + hi * 16;
        const float* mp = Mu + u * IDIM + frag * 32 + hi * 16;
#pragma unroll
        for (int j = 0; j < 16; ++j) {
            float s = sp[j] * LOG2E;
            sgr[frag * 16 + j] = s;
            c0r[frag * 16 + j] = s * mp[j];
        }
    }

    // Constant A-matrix fragments (16x32 f16 layout: lane holds row M = lane%16;
    // k order chosen here, same as B: half j -> K = 16*hi + j).
    v16h afrag[2];
#pragma unroll
    for (int frag = 0; frag < 2; ++frag) {
#pragma unroll
        for (int j = 0; j < 16; ++j) {
            int k = j + 16 * hi;
            float av = Aw[u * IDIM + frag * 32 + k];
            _Float16 h = (_Float16)0.0f;
            if (n == 0) h = (_Float16)1.0f;        // row 0: ones -> fs
            else if (n == 1) h = (_Float16)av;     // row 1: A[u,:] -> sum(A*f)
            afrag[frag][j] = h;
        }
    }

    float x = X0[u];   // scan state (carried on lanes n<8)

    for (int t0 = 0; t0 < TSTEPS; t0 += 2) {
        const float* ip = inp + ((size_t)bb * TSTEPS + (t0 + tsel)) * IDIM + hi * 16;
        __builtin_prefetch(ip + 2 * IDIM, 0, 1);   // next t-pair

        v8f c = {0.f, 0.f, 0.f, 0.f, 0.f, 0.f, 0.f, 0.f};
#pragma unroll
        for (int frag = 0; frag < 2; ++frag) {
            v16h bf;   // B-matrix fragment: half j <-> d = frag*32 + hi*16 + j
#pragma unroll
            for (int j = 0; j < 16; j += 2) {
                float z0 = ip[frag * 32 + j];
                float z1 = ip[frag * 32 + j + 1];
                float w0 = __builtin_fmaf(-sgr[frag * 16 + j],     z0, c0r[frag * 16 + j]);
                float w1 = __builtin_fmaf(-sgr[frag * 16 + j + 1], z1, c0r[frag * 16 + j + 1]);
                float f0 = __builtin_amdgcn_rcpf(1.0f + __builtin_amdgcn_exp2f(w0));
                float f1 = __builtin_amdgcn_rcpf(1.0f + __builtin_amdgcn_exp2f(w1));
                v2h p = __builtin_bit_cast(v2h, __builtin_amdgcn_cvt_pkrtz(f0, f1));
                bf[j]     = p[0];
                bf[j + 1] = p[1];
            }
            // (neg_a, A, neg_b, B, c_mod, C, reuse_a, reuse_b)
            c = __builtin_amdgcn_wmma_f32_16x16x32_f16(false, afrag[frag],
                                                       false, bf,
                                                       (short)0, c, false, false);
        }

        // c[0] = fs, c[1] = sum(A*f) for this lane's (b, t0+tsel)
        float fs = c[0];
        float s  = c[1] * __builtin_amdgcn_rcpf(1.0f + fs);
        float a  = __builtin_amdgcn_exp2f(-(1.0f + fs) * LOG2E); // exp(-1-fs)

        // timestep-pair scan: lanes 8..15 carry the (a,s) of t0+1
        float a1 = __shfl(a, lane | 8, 32);
        float s1 = __shfl(s, lane | 8, 32);
        float y0 = __builtin_fmaf(a,  x  - s,  s );   // x_{t0}   (valid on n<8)
        float y1 = __builtin_fmaf(a1, y0 - s1, s1);   // x_{t0+1} (valid on n<8)
        float yo = __shfl(y1, lane & 7, 32);          // move x_{t0+1} to lanes 8..15
        float y  = (n < 8) ? y0 : yo;

        if (lane < 16) {  // divergence only around the store; WMMA runs full-EXEC
            out[((size_t)bb * TSTEPS + (t0 + tsel)) * UNITS + u] = y;
        }
        x = y1;
    }
}

extern "C" void kernel_launch(void* const* d_in, const int* in_sizes, int n_in,
                              void* d_out, int out_size, void* d_ws, size_t ws_size,
                              hipStream_t stream) {
    (void)in_sizes; (void)n_in; (void)d_ws; (void)ws_size; (void)out_size;
    const float* inp = (const float*)d_in[0];  // [128,32,32,64] -> [B,T,D]
    const float* Aw  = (const float*)d_in[1];  // [256,64]
    const float* Sg  = (const float*)d_in[2];  // [256,64]
    const float* Mu  = (const float*)d_in[3];  // [256,64]
    const float* X0  = (const float*)d_in[4];  // [256]
    float* out = (float*)d_out;                // [B,T,U]

    // 16 b-tiles * 256 units = 4096 waves = 131072 threads; 8 waves/block
    dim3 grid(512), block(256);
    ltc_scan_wmma<<<grid, block, 0, stream>>>(inp, Aw, Sg, Mu, X0, out);
}
